// MultiHeadLatentAttentionNaive_26001732010569
// MI455X (gfx1250) — matrix operand order repaired and checked by reading, not compiled, not run on hardware
//
#include <hip/hip_runtime.h>

typedef __attribute__((ext_vector_type(16))) __bf16 v16bf;
typedef __attribute__((ext_vector_type(8)))  __bf16 v8bf;
typedef __attribute__((ext_vector_type(2)))  __bf16 v2bf;
typedef __attribute__((ext_vector_type(8)))  float  v8f;

// ---- fp32 -> bf16 conversion helpers -------------------------------------
// Prefer the hardware packed converter (single VALU op, RNE); fall back to a
// manual round-to-nearest-even bit path.
__device__ __forceinline__ unsigned int f2bf_manual(float f) {
    unsigned int u = __float_as_uint(f);
    u += 0x7FFFu + ((u >> 16) & 1u);
    return u >> 16;
}

#if __has_builtin(__builtin_amdgcn_cvt_pk_bf16_f32)
__device__ __forceinline__ unsigned int pk2bf(float a, float b) {
    auto r = __builtin_amdgcn_cvt_pk_bf16_f32(a, b);
    unsigned int u = 0;
    __builtin_memcpy(&u, &r, sizeof(r) <= 4 ? sizeof(r) : 4);
    return u;
}
#else
__device__ __forceinline__ unsigned int pk2bf(float a, float b) {
    return f2bf_manual(a) | (f2bf_manual(b) << 16);
}
#endif

__device__ __forceinline__ unsigned short f2bf(float f) {
    return (unsigned short)(pk2bf(f, f) & 0xFFFFu);
}

// A fragment (16x32 bf16, MxK), source: row-major [m][k] LDS tile, `base` at [m0][k0].
// lane<16: K=0..15 ; lane>=16: K=16..31 (contiguous per lane).
__device__ __forceinline__ v16bf load_a_frag(const unsigned short* base, int stride, int lane) {
    int hf = (lane >> 4) & 1;
    int r  = lane & 15;
    const unsigned short* p = base + (size_t)r * stride + hf * 16;
    v8bf lo = *(const v8bf*)(p);
    v8bf hi = *(const v8bf*)(p + 8);
    return __builtin_shufflevector(lo, hi, 0,1,2,3,4,5,6,7,8,9,10,11,12,13,14,15);
}

// B fragment (32x16 bf16, KxN), source: LDS tile stored [n][k], `base` at [n0][k0].
// lane<16: K=0..15 ; lane>=16: K=16..31 (contiguous per lane).
__device__ __forceinline__ v16bf load_b_frag(const unsigned short* base, int stride, int lane) {
    int hf = (lane >> 4) & 1;
    int c  = lane & 15;
    const unsigned short* p = base + (size_t)c * stride + hf * 16;
    v8bf lo = *(const v8bf*)(p);
    v8bf hi = *(const v8bf*)(p + 8);
    return __builtin_shufflevector(lo, hi, 0,1,2,3,4,5,6,7,8,9,10,11,12,13,14,15);
}

__device__ __forceinline__ v8f wmma_bf16(v16bf a, v16bf b, v8f c) {
    return __builtin_amdgcn_wmma_f32_16x16x32_bf16(false, a, false, b, (short)0, c, false, false);
}

// ---------------------------------------------------------------------------
// Generic GEMM: C[M,N] = A[M,K] @ B[K,N] (+ bias[N]), fp32 in/out, bf16 WMMA.
// Block 256 threads = 8 waves, tile 128x128, BK=32. Wave grid 2x4, wave tile 64x32.
// Requires M%128==0, N%128==0, K%32==0.
// ---------------------------------------------------------------------------
__global__ __launch_bounds__(256)
void mla_gemm_bf16(const float* __restrict__ A, const float* __restrict__ Bm,
                   const float* __restrict__ bias, float* __restrict__ C,
                   int M, int N, int K) {
    __shared__ __align__(16) unsigned short As[128][40];   // [m][k]
    __shared__ __align__(16) unsigned short Bs[128][40];   // [n][k] (transposed)

    const int tid  = threadIdx.x;
    const int lane = tid & 31;
    const int wave = tid >> 5;
    const int wm   = wave >> 2;         // 0..1
    const int wn   = wave & 3;          // 0..3
    const int m0   = blockIdx.y * 128;
    const int n0   = blockIdx.x * 128;

    const int a_rowb = tid >> 3;          // 0..31
    const int a_col  = (tid & 7) * 4;     // 0,4,..,28

    v8f acc[4][2];
    v8f vz = {0.f,0.f,0.f,0.f,0.f,0.f,0.f,0.f};
#pragma unroll
    for (int mi = 0; mi < 4; ++mi)
#pragma unroll
        for (int ni = 0; ni < 2; ++ni) acc[mi][ni] = vz;

    for (int k0 = 0; k0 < K; k0 += 32) {
        // Stage A tile (128x32): float4 global load -> packed b64 LDS store
#pragma unroll
        for (int p = 0; p < 4; ++p) {
            int row = a_rowb + p * 32;
            float4 v = *(const float4*)(A + (size_t)(m0 + row) * K + k0 + a_col);
            uint2 u; u.x = pk2bf(v.x, v.y); u.y = pk2bf(v.z, v.w);
            *(uint2*)&As[row][a_col] = u;
        }
        // Stage B tile (32x128) transposed into [n][k]:
        // thread handles 4 consecutive k at one n (coalesced b32 loads over n),
        // one packed b64 LDS store.
#pragma unroll
        for (int p = 0; p < 4; ++p) {
            int c  = tid + p * 256;        // 0..1023
            int n  = c & 127;
            int kq = (c >> 7) * 4;         // 0,4,..,28
            const float* bp = Bm + (size_t)(k0 + kq) * N + n0 + n;
            float b0 = bp[0];
            float b1 = bp[(size_t)N];
            float b2 = bp[(size_t)2 * N];
            float b3 = bp[(size_t)3 * N];
            uint2 u; u.x = pk2bf(b0, b1); u.y = pk2bf(b2, b3);
            *(uint2*)&Bs[n][kq] = u;
        }
        // Prefetch next k-tile while this one is consumed (global_prefetch_b8)
        if (k0 + 32 < K) {
            __builtin_prefetch(A + (size_t)(m0 + a_rowb) * K + k0 + 32 + a_col, 0, 1);
            __builtin_prefetch(Bm + (size_t)(k0 + 32 + ((tid >> 7) * 4)) * N + n0 + (tid & 127), 0, 1);
        }
        __syncthreads();

        v16bf af[4];
#pragma unroll
        for (int mi = 0; mi < 4; ++mi)
            af[mi] = load_a_frag(&As[wm * 64 + mi * 16][0], 40, lane);
#pragma unroll
        for (int ni = 0; ni < 2; ++ni) {
            v16bf bf = load_b_frag(&Bs[wn * 32 + ni * 16][0], 40, lane);
#pragma unroll
            for (int mi = 0; mi < 4; ++mi)
                acc[mi][ni] = wmma_bf16(af[mi], bf, acc[mi][ni]);
        }
        __syncthreads();
    }

    // Epilogue: C-frag layout row = r + 8*(lane>=16), col = lane&15
    const int hf  = lane >> 4;
    const int cid = lane & 15;
#pragma unroll
    for (int mi = 0; mi < 4; ++mi)
#pragma unroll
        for (int ni = 0; ni < 2; ++ni) {
            int col = n0 + wn * 32 + ni * 16 + cid;
            float bb = bias ? bias[col] : 0.f;
#pragma unroll
            for (int r = 0; r < 8; ++r) {
                int row = m0 + wm * 64 + mi * 16 + r + hf * 8;
                C[(size_t)row * N + col] = acc[mi][ni][r] + bb;
            }
        }
}

// ---------------------------------------------------------------------------
// Flash attention, hd=64. Block = 128 thr (4 waves), 64 query rows per block,
// one wave = 16 query rows. Streams 64-key K/V tiles, causal, online softmax.
// Grid: (S/64, H, B). q: (B,S,D) fp32; kv: (B,S,2D) fp32 (k | v); ctx: (B,S,D).
// ---------------------------------------------------------------------------
__global__ __launch_bounds__(128)
void mla_flash_attn(const float* __restrict__ Q, const float* __restrict__ KV,
                    const int* __restrict__ offp, float* __restrict__ ctx,
                    int S, int D, int H) {
    const int HD = 64;
    __shared__ __align__(16) unsigned short Qs[64][72];     // [q][d], pre-scaled
    __shared__ __align__(16) unsigned short Ks[64][72];     // [key][d]  -> B frag [n=key][k=d]
    __shared__ __align__(16) unsigned short Vs[64][72];     // [d][key]  -> B frag [n=d][k=key]
    __shared__ __align__(16) unsigned short Ps[4][16][72];  // per-wave P [q][key]

    const int tid  = threadIdx.x;
    const int lane = tid & 31;
    const int wave = tid >> 5;
    const int hf   = lane >> 4;
    const int cid  = lane & 15;

    const int q0 = blockIdx.x * 64;
    const int h  = blockIdx.y;
    const int b  = blockIdx.z;
    const int off = *offp;

    const float* Qg = Q  + (size_t)b * S * D       + (size_t)h * HD;
    const float* Kg = KV + (size_t)b * S * (2 * D) + (size_t)h * HD;
    const float* Vg = Kg + D;
    const float scale = 0.125f;   // 1/sqrt(64)
    const size_t ldkv = 2 * D;

    // Load + scale Q tile (64x64): float4 load -> packed b64 LDS store
#pragma unroll
    for (int p = 0; p < 8; ++p) {
        int c = tid + p * 128;
        int row = c >> 4, col = (c & 15) * 4;
        float4 v = *(const float4*)(Qg + (size_t)(q0 + row) * D + col);
        uint2 u; u.x = pk2bf(v.x * scale, v.y * scale); u.y = pk2bf(v.z * scale, v.w * scale);
        *(uint2*)&Qs[row][col] = u;
    }

    v8f vz = {0.f,0.f,0.f,0.f,0.f,0.f,0.f,0.f};
    v8f oacc[4];
#pragma unroll
    for (int ni = 0; ni < 4; ++ni) oacc[ni] = vz;
    float mrow[8], lrow[8];
#pragma unroll
    for (int r = 0; r < 8; ++r) { mrow[r] = -1e30f; lrow[r] = 0.f; }

    int kend = q0 + 64 + off;
    if (kend > S) kend = S;
    const int ntiles = (kend + 63) >> 6;

    for (int t = 0; t < ntiles; ++t) {
        const int j0 = t * 64;
        __syncthreads();
        // Stage K [key][d]: float4 loads along d, packed b64 stores
#pragma unroll
        for (int p = 0; p < 8; ++p) {
            int c = tid + p * 128;
            int row = c >> 4, col = (c & 15) * 4;
            float4 kv4 = *(const float4*)(Kg + (size_t)(j0 + row) * ldkv + col);
            uint2 u; u.x = pk2bf(kv4.x, kv4.y); u.y = pk2bf(kv4.z, kv4.w);
            *(uint2*)&Ks[row][col] = u;
        }
        // Stage V transposed [d][key]: 4 key-consecutive b32 loads (coalesced
        // over d), one packed b64 store per thread-chunk.
#pragma unroll
        for (int p = 0; p < 8; ++p) {
            int c = tid + p * 128;
            int d  = c & 63;
            int kq = (c >> 6) * 4;          // 0,4,..,60
            const float* vp = Vg + (size_t)(j0 + kq) * ldkv + d;
            float v0 = vp[0];
            float v1 = vp[ldkv];
            float v2 = vp[2 * ldkv];
            float v3 = vp[3 * ldkv];
            uint2 u; u.x = pk2bf(v0, v1); u.y = pk2bf(v2, v3);
            *(uint2*)&Vs[d][kq] = u;
        }
        // Prefetch next key tile (global_prefetch_b8)
        if (t + 1 < ntiles) {
            int row = tid >> 4, col = (tid & 15) * 4;
            __builtin_prefetch(Kg + (size_t)(j0 + 64 + row) * ldkv + col, 0, 1);
            __builtin_prefetch(Vg + (size_t)(j0 + 64 + row) * ldkv + col, 0, 1);
        }
        __syncthreads();

        // S = Q @ K^T  (16 x 64 per wave)
        v8f sacc[4];
#pragma unroll
        for (int ni = 0; ni < 4; ++ni) sacc[ni] = vz;
#pragma unroll
        for (int kk = 0; kk < 64; kk += 32) {
            v16bf aq = load_a_frag(&Qs[wave * 16][kk], 72, lane);
#pragma unroll
            for (int ni = 0; ni < 4; ++ni) {
                v16bf bk = load_b_frag(&Ks[ni * 16][kk], 72, lane);
                sacc[ni] = wmma_bf16(aq, bk, sacc[ni]);
            }
        }

        // Causal mask + online softmax (rows r + 8*hf, cols across 16 lanes)
#pragma unroll
        for (int r = 0; r < 8; ++r) {
            int qpos = q0 + wave * 16 + r + hf * 8 + off;
            float rowmax = -1e30f;
#pragma unroll
            for (int ni = 0; ni < 4; ++ni) {
                int key = j0 + ni * 16 + cid;
                float s = sacc[ni][r];
                if (key > qpos) s = -1e30f;
                sacc[ni][r] = s;
                rowmax = fmaxf(rowmax, s);
            }
            rowmax = fmaxf(rowmax, __shfl_xor(rowmax, 1));
            rowmax = fmaxf(rowmax, __shfl_xor(rowmax, 2));
            rowmax = fmaxf(rowmax, __shfl_xor(rowmax, 4));
            rowmax = fmaxf(rowmax, __shfl_xor(rowmax, 8));
            float mnew = fmaxf(mrow[r], rowmax);
            float corr = __expf(mrow[r] - mnew);
            float rowsum = 0.f;
#pragma unroll
            for (int ni = 0; ni < 4; ++ni) {
                float pv = __expf(sacc[ni][r] - mnew);
                rowsum += pv;
                Ps[wave][r + hf * 8][ni * 16 + cid] = f2bf(pv);
            }
            rowsum += __shfl_xor(rowsum, 1);
            rowsum += __shfl_xor(rowsum, 2);
            rowsum += __shfl_xor(rowsum, 4);
            rowsum += __shfl_xor(rowsum, 8);
            lrow[r] = lrow[r] * corr + rowsum;
            mrow[r] = mnew;
#pragma unroll
            for (int ni = 0; ni < 4; ++ni) oacc[ni][r] = oacc[ni][r] * corr;
        }
        __syncthreads();

        // O += P @ V   (A = Ps [q][key], B = Vs [n=d][k=key])
#pragma unroll
        for (int kk = 0; kk < 64; kk += 32) {
            v16bf ap = load_a_frag(&Ps[wave][0][kk], 72, lane);
#pragma unroll
            for (int ni = 0; ni < 4; ++ni) {
                v16bf bv = load_b_frag(&Vs[ni * 16][kk], 72, lane);
                oacc[ni] = wmma_bf16(ap, bv, oacc[ni]);
            }
        }
    }

    // Normalize and write ctx
    float* Cg = ctx + (size_t)b * S * D + (size_t)h * HD;
#pragma unroll
    for (int r = 0; r < 8; ++r) {
        float inv = 1.f / lrow[r];
        int row = q0 + wave * 16 + r + hf * 8;
#pragma unroll
        for (int ni = 0; ni < 4; ++ni)
            Cg[(size_t)row * D + ni * 16 + cid] = oacc[ni][r] * inv;
    }
}

// ---------------------------------------------------------------------------
extern "C" void kernel_launch(void* const* d_in, const int* in_sizes, int n_in,
                              void* d_out, int out_size, void* d_ws, size_t ws_size,
                              hipStream_t stream) {
    (void)in_sizes; (void)n_in; (void)out_size; (void)ws_size;
    const float* x    = (const float*)d_in[0];
    const int*   offp = (const int*)  d_in[1];
    const float* Wq   = (const float*)d_in[2];
    const float* Wdkv = (const float*)d_in[3];
    const float* Wukv = (const float*)d_in[4];
    const float* Wo   = (const float*)d_in[5];
    const float* bo   = (const float*)d_in[6];
    float* out = (float*)d_out;

    const int B = 2, S = 2048, D = 1024, L = 128, H = 16;
    const int M = B * S;   // 4096 token rows

    float* qb  = (float*)d_ws;                       // M*D
    float* lat = qb  + (size_t)M * D;                // M*L
    float* kv  = lat + (size_t)M * L;                // M*2D
    float* ctx = kv  + (size_t)M * 2 * D;            // M*D

    // q = x @ Wq          (4096x1024 @ 1024x1024)
    mla_gemm_bf16<<<dim3(D / 128, M / 128), 256, 0, stream>>>(x, Wq, nullptr, qb, M, D, D);
    // latent = x @ Wdkv   (4096x1024 @ 1024x128)
    mla_gemm_bf16<<<dim3(L / 128, M / 128), 256, 0, stream>>>(x, Wdkv, nullptr, lat, M, L, D);
    // kv = latent @ Wukv  (4096x128 @ 128x2048)
    mla_gemm_bf16<<<dim3((2 * D) / 128, M / 128), 256, 0, stream>>>(lat, Wukv, nullptr, kv, M, 2 * D, L);
    // attention -> ctx
    mla_flash_attn<<<dim3(S / 64, H, B), 128, 0, stream>>>(qb, kv, offp, ctx, S, D, H);
    // out = ctx @ Wo + bo
    mla_gemm_bf16<<<dim3(D / 128, M / 128), 256, 0, stream>>>(ctx, Wo, bo, out, M, D, D);
}
